// LongcatFlashForCausalLM_44882408243533
// MI455X (gfx1250) — matrix operand order repaired and checked by reading, not compiled, not run on hardware
//
#include <hip/hip_runtime.h>
#include <hip/hip_bf16.h>
#include <math.h>

// ---------------------------------------------------------------------------
// LongCat-Flash block for MI455X (gfx1250), wave32 + WMMA bf16.
// fp32 in memory -> float4 global loads -> bf16 round (+0x8000, perm-pack) ->
// LDS (register double-buffered, XOR-swizzled B, peeled last iter) ->
// v_wmma_f32_16x16x32_bf16 -> fp32 out.  MoE is gathered top-2.
// ---------------------------------------------------------------------------

#define T_N   2048
#define D_N   2048
#define H_N   16
#define DN_N  128
#define DR_N  64
#define DV_N  128
#define QL_N  768
#define KL_N  512
#define FF_N  8192
#define E_N   32
#define EF_N  1024
#define EPS_F 1e-6f

typedef __attribute__((ext_vector_type(16))) __bf16 v16bf;
typedef __attribute__((ext_vector_type(8)))  float  v8f;

union FragAB { uint4 q[2]; v16bf v; };

// round float to bf16 in bits [31:16] (round-to-nearest, ties away)
__device__ __forceinline__ unsigned int rnbf(float f) {
  return __float_as_uint(f) + 0x8000u;
}
// pack two floats into two bf16 (lo -> [15:0], hi -> [31:16]) : 3 VALU
__device__ __forceinline__ unsigned int pack2bf(float lo, float hi) {
  return __builtin_amdgcn_perm(rnbf(hi), rnbf(lo), 0x07060302u);
}

// ---------------------------------------------------------------------------
// Generic batched WMMA GEMM:  C[z] (+)= alpha * rowscale * A[z] x B[z]
//   A fp32 row-major (lda), optional row gather a_idx (hoisted per-thread)
//   B fp32, element (k,n) at B + k*brs + n  (n contiguous in all call sites)
//   C fp32, optional row scatter / per-row scale / accumulate
//   M overridable by device pointer (gathered-MoE expert counts)
// Block tile 128x128, 8 waves (2x4), wave tile 64x32 -> 8 WMMA accums.
// K multiple of 32 (true for every GEMM here).  Double-buffered staging.
// ---------------------------------------------------------------------------
struct GemmP {
  const float* A; long lda; long abat;
  const float* B; long brs; long bbat;
  float* C; long ldc; long cbat;
  int M, N, K;
  float alpha;
  const int*   a_idx;
  const int*   c_idx;
  const float* c_scale;
  const int*   m_dev;
  int accum;
};

#define LSTR 40   // ushort stride: 80B rows -> 16B-aligned b128 chunks

__global__ __launch_bounds__(256) void gemm_bf16_wmma(GemmP p) {
  const int tid  = threadIdx.x;
  const int lane = tid & 31;
  const int wid  = tid >> 5;
  const int z    = blockIdx.z;
  const int bm   = blockIdx.y * 128;
  const int bn   = blockIdx.x * 128;

  const int Meff = p.m_dev ? *p.m_dev : p.M;
  if (bm >= Meff) return;

  const float* Ab = p.A + (long)z * p.abat;
  const float* Bb = p.B + (long)z * p.bbat;
  float*       Cb = p.C + (long)z * p.cbat;

  __shared__ __align__(16) unsigned short lA[128 * LSTR];
  __shared__ __align__(16) unsigned short lB[128 * LSTR];   // [n][k^swz]

  // ---- per-thread staging setup (hoisted out of the K loop) ----
  // A: thread owns row m = tid>>1, k-range akof..akof+15
  const int am   = tid >> 1;
  const int akof = (tid & 1) * 16;
  const float* arow = nullptr;
  {
    int gm = bm + am;
    if (gm < Meff) {
      long row = p.a_idx ? (long)p.a_idx[gm] : (long)gm;
      arow = Ab + row * p.lda + akof;
    }
  }
  // B: thread owns k = tid>>3, n-range bnof..bnof+15
  const int bk   = tid >> 3;
  const int bnof = (tid & 7) * 16;
  const int bcol = bk ^ (((bnof >> 4) & 3) << 3);   // XOR bank swizzle
  const float* brow = Bb + (long)bk * p.brs + bn + bnof;
  const bool nfull = (bn + 128 <= p.N);

  float4 ra[4], rb[4];
  const float4 z4 = make_float4(0.f, 0.f, 0.f, 0.f);

  auto load_regs = [&](int k0) {
    if (arow) {
      const float4* pa = (const float4*)(arow + k0);
      ra[0] = pa[0]; ra[1] = pa[1]; ra[2] = pa[2]; ra[3] = pa[3];
    } else {
      ra[0] = z4; ra[1] = z4; ra[2] = z4; ra[3] = z4;
    }
    if (nfull) {
      const float4* pb = (const float4*)(brow + (long)k0 * p.brs);
      rb[0] = pb[0]; rb[1] = pb[1]; rb[2] = pb[2]; rb[3] = pb[3];
    } else {
      const float* pb = brow + (long)k0 * p.brs;
      float t[16];
      #pragma unroll
      for (int j = 0; j < 16; ++j)
        t[j] = (bn + bnof + j < p.N) ? pb[j] : 0.f;
      rb[0] = make_float4(t[0], t[1], t[2], t[3]);
      rb[1] = make_float4(t[4], t[5], t[6], t[7]);
      rb[2] = make_float4(t[8], t[9], t[10], t[11]);
      rb[3] = make_float4(t[12], t[13], t[14], t[15]);
    }
  };

  auto store_lds = [&]() {
    // A: 16 floats -> 8 packed dwords -> 2x ds_store_b128
    uint4 u0, u1;
    u0.x = pack2bf(ra[0].x, ra[0].y); u0.y = pack2bf(ra[0].z, ra[0].w);
    u0.z = pack2bf(ra[1].x, ra[1].y); u0.w = pack2bf(ra[1].z, ra[1].w);
    u1.x = pack2bf(ra[2].x, ra[2].y); u1.y = pack2bf(ra[2].z, ra[2].w);
    u1.z = pack2bf(ra[3].x, ra[3].y); u1.w = pack2bf(ra[3].z, ra[3].w);
    uint4* pa = (uint4*)&lA[am * LSTR + akof];
    pa[0] = u0; pa[1] = u1;
    // B transposed (swizzled column bcol): round then store high half
    // (compiler lowers to v_add + ds_store_b16_d16_hi, no pack ops)
    const float* f = (const float*)rb;
    #pragma unroll
    for (int j = 0; j < 16; ++j) {
      unsigned int r = rnbf(f[j]);
      lB[(bnof + j) * LSTR + bcol] = (unsigned short)(r >> 16);
    }
  };

  const int mw = (wid >> 2) * 64;
  const int nw = (wid & 3) * 32;

  v8f acc[4][2];
  #pragma unroll
  for (int i = 0; i < 4; ++i)
    #pragma unroll
    for (int j = 0; j < 2; ++j)
      #pragma unroll
      for (int e = 0; e < 8; ++e) acc[i][j][e] = 0.f;

  auto compute_tile = [&]() {
    FragAB fa[4], fb[2];
    const int akb = (lane < 16) ? 0 : 16;
    #pragma unroll
    for (int mt = 0; mt < 4; ++mt) {
      const unsigned short* pA = &lA[(mw + mt * 16 + (lane & 15)) * LSTR + akb];
      fa[mt].q[0] = *(const uint4*)(pA);       // K = akb..akb+7
      fa[mt].q[1] = *(const uint4*)(pA + 8);   // K = akb+8..akb+15
    }
    const int bkb = (lane < 16) ? 0 : 16;
    #pragma unroll
    for (int nt = 0; nt < 2; ++nt) {
      int n  = nw + nt * 16 + (lane & 15);
      int xb = ((n >> 4) & 3) << 3;
      const unsigned short* pB = &lB[n * LSTR];
      fb[nt].q[0] = *(const uint4*)(pB + (bkb ^ xb));        // K = bkb..bkb+7
      fb[nt].q[1] = *(const uint4*)(pB + ((bkb + 8) ^ xb));  // K = bkb+8..bkb+15
    }
    #pragma unroll
    for (int mt = 0; mt < 4; ++mt)
      #pragma unroll
      for (int nt = 0; nt < 2; ++nt)
        acc[mt][nt] = __builtin_amdgcn_wmma_f32_16x16x32_bf16(
            false, fa[mt].v, false, fb[nt].v, (short)0, acc[mt][nt], false, false);
  };

  load_regs(0);
  store_lds();

  int k0 = 0;
  for (; k0 + 32 < p.K; k0 += 32) {        // steady state: always prefetches
    __syncthreads();
    load_regs(k0 + 32);
    compute_tile();
    __syncthreads();
    store_lds();
  }
  __syncthreads();                          // peeled last tile
  compute_tile();

  #pragma unroll
  for (int mt = 0; mt < 4; ++mt) {
    #pragma unroll
    for (int v = 0; v < 8; ++v) {
      int r = bm + mw + mt * 16 + ((lane < 16) ? v : v + 8);
      if (r >= Meff) continue;
      long orow = p.c_idx ? (long)p.c_idx[r] : (long)r;
      float rs  = p.c_scale ? p.c_scale[r] : 1.f;
      #pragma unroll
      for (int nt = 0; nt < 2; ++nt) {
        int c = bn + nw + nt * 16 + (lane & 15);
        if (c >= p.N) continue;
        float val = p.alpha * acc[mt][nt][v] * rs;
        float* cp = Cb + orow * p.ldc + c;
        if (p.accum) val += *cp;
        *cp = val;
      }
    }
  }
}

// ---------------------------------------------------------------------------
// Elementwise / normalization kernels
// ---------------------------------------------------------------------------
__global__ void rms_kernel(const float* __restrict__ x, long ldx,
                           const float* __restrict__ w,
                           float* __restrict__ out, long ldo, int d) {
  int row = blockIdx.x;
  const float* xr = x + (long)row * ldx;
  float ss = 0.f;
  for (int i = threadIdx.x; i < d; i += 256) { float v = xr[i]; ss += v * v; }
  __shared__ float red[256];
  red[threadIdx.x] = ss; __syncthreads();
  for (int s = 128; s > 0; s >>= 1) {
    if (threadIdx.x < s) red[threadIdx.x] += red[threadIdx.x + s];
    __syncthreads();
  }
  float inv = rsqrtf(red[0] / (float)d + EPS_F);
  for (int i = threadIdx.x; i < d; i += 256)
    out[(long)row * ldo + i] = xr[i] * inv * w[i];
}

__global__ void add_rms_kernel(const float* __restrict__ a, const float* __restrict__ b,
                               const float* __restrict__ w,
                               float* __restrict__ res_out, float* __restrict__ norm_out,
                               int d) {
  int row = blockIdx.x; long base = (long)row * d;
  float ss = 0.f;
  for (int i = threadIdx.x; i < d; i += 256) {
    float r = a[base + i] + b[base + i];
    res_out[base + i] = r;
    ss += r * r;
  }
  __shared__ float red[256];
  red[threadIdx.x] = ss; __syncthreads();
  for (int s = 128; s > 0; s >>= 1) {
    if (threadIdx.x < s) red[threadIdx.x] += red[threadIdx.x + s];
    __syncthreads();
  }
  float inv = rsqrtf(red[0] / (float)d + EPS_F);
  for (int i = threadIdx.x; i < d; i += 256)
    norm_out[base + i] = res_out[base + i] * inv * w[i];
}

// RoPE on q rope-halves, in place.  q layout (T, H, 192), rope at [128..192)
__global__ void rope_q_kernel(float* __restrict__ q, const int* __restrict__ pos) {
  int gid = blockIdx.x * 256 + threadIdx.x;
  int pair = gid >> 5, i = gid & 31;
  if (pair >= T_N * H_N) return;
  int t = pair / H_N, h = pair % H_N;
  float p   = (float)pos[t];
  float inv = __expf(-logf(10000.f) * (2.f * (float)i / (float)DR_N));
  float f = p * inv, c = __cosf(f), s = __sinf(f);
  float* base = q + (long)t * (H_N * 192) + h * 192 + DN_N;
  float x1 = base[i], x2 = base[32 + i];
  base[i]      = x1 * c - x2 * s;
  base[32 + i] = x2 * c + x1 * s;
}

// Assemble K^T per head:  khT[h][d][t] ; d<128 -> k_nope (kvb), d>=128 -> roped rope
__global__ void khT_kernel(const float* __restrict__ kvb, const float* __restrict__ kv,
                           const int* __restrict__ pos, float* __restrict__ khT) {
  int t = blockIdx.x;
  __shared__ float kr[64];
  if (threadIdx.x < 32) {
    int i = threadIdx.x;
    float p   = (float)pos[t];
    float inv = __expf(-logf(10000.f) * (2.f * (float)i / (float)DR_N));
    float f = p * inv, c = __cosf(f), s = __sinf(f);
    float x1 = kv[(long)t * (KL_N + DR_N) + KL_N + i];
    float x2 = kv[(long)t * (KL_N + DR_N) + KL_N + 32 + i];
    kr[i]      = x1 * c - x2 * s;
    kr[i + 32] = x2 * c + x1 * s;
  }
  __syncthreads();
  for (int idx = threadIdx.x; idx < H_N * 192; idx += 256) {
    int h = idx / 192, d = idx % 192;
    float v = (d < DN_N) ? kvb[(long)t * (H_N * 256) + h * 256 + d] : kr[d - DN_N];
    khT[((long)h * 192 + d) * T_N + t] = v;
  }
}

// causal softmax over scores[h][t][:]  (T == 8*256)
__global__ void softmax_causal_kernel(float* __restrict__ sc, const int* __restrict__ pos) {
  int t = blockIdx.x, h = blockIdx.y;
  float* row = sc + ((long)h * T_N + t) * T_N;
  int pt = pos[t];
  float loc[8];
  float mx = -1e30f;
  #pragma unroll
  for (int j = 0; j < 8; ++j) {
    int s = threadIdx.x + j * 256;
    float v = (pos[s] <= pt) ? row[s] : -1e9f;
    loc[j] = v; mx = fmaxf(mx, v);
  }
  __shared__ float red[256];
  red[threadIdx.x] = mx; __syncthreads();
  for (int s = 128; s > 0; s >>= 1) {
    if (threadIdx.x < s) red[threadIdx.x] = fmaxf(red[threadIdx.x], red[threadIdx.x + s]);
    __syncthreads();
  }
  mx = red[0]; __syncthreads();
  float sum = 0.f;
  #pragma unroll
  for (int j = 0; j < 8; ++j) { loc[j] = __expf(loc[j] - mx); sum += loc[j]; }
  red[threadIdx.x] = sum; __syncthreads();
  for (int s = 128; s > 0; s >>= 1) {
    if (threadIdx.x < s) red[threadIdx.x] += red[threadIdx.x + s];
    __syncthreads();
  }
  float inv = 1.f / red[0];
  #pragma unroll
  for (int j = 0; j < 8; ++j) row[threadIdx.x + j * 256] = loc[j] * inv;
}

// prod[t,f] = silu(gu[t,f]) * gu[t,FF+f]
__global__ void silu_mul_gu_kernel(const float* __restrict__ gu, float* __restrict__ prod) {
  long i = (long)blockIdx.x * 256 + threadIdx.x;
  if (i >= (long)T_N * FF_N) return;
  long t = i / FF_N, f = i % FF_N;
  float x = gu[t * (2 * FF_N) + f];
  float y = gu[t * (2 * FF_N) + FF_N + f];
  prod[i] = (x / (1.f + __expf(-x))) * y;
}

// gathered MoE gate fuse: g = silu(g)*u over count*EF elements (device count)
__global__ void silu_mul_cnt_kernel(float* __restrict__ g, const float* __restrict__ u,
                                    const int* __restrict__ cnt) {
  long i = (long)blockIdx.x * 256 + threadIdx.x;
  long n = (long)(*cnt) * EF_N;
  if (i >= n) return;
  float x = g[i];
  g[i] = (x / (1.f + __expf(-x))) * u[i];
}

// deterministic top-2 of router logits (strict '>' keeps lowest index, like lax.top_k)
__global__ void top2_kernel(const float* __restrict__ logits,
                            int* __restrict__ ti, float* __restrict__ tw) {
  int t = blockIdx.x * 256 + threadIdx.x;
  if (t >= T_N) return;
  const float* l = logits + (long)t * E_N;
  int i0 = 0; float b0 = l[0];
  for (int e = 1; e < E_N; ++e) { float v = l[e]; if (v > b0) { b0 = v; i0 = e; } }
  int i1 = (i0 == 0) ? 1 : 0; float b1 = l[i1];
  for (int e = 0; e < E_N; ++e) {
    if (e == i0) continue;
    float v = l[e]; if (v > b1) { b1 = v; i1 = e; }
  }
  float w0 = 1.f / (1.f + __expf(b1 - b0));   // p0/(p0+p1)
  ti[t * 2] = i0; ti[t * 2 + 1] = i1;
  tw[t * 2] = w0; tw[t * 2 + 1] = 1.f - w0;
}

// one thread per expert: sequential, atomics-free, deterministic token lists
__global__ void route_build_kernel(const int* __restrict__ ti, const float* __restrict__ tw,
                                   int* __restrict__ rows, float* __restrict__ scale,
                                   int* __restrict__ counts) {
  int e = threadIdx.x;
  if (e >= E_N) return;
  int c = 0;
  for (int t = 0; t < T_N; ++t) {
    if (ti[t * 2] == e)          { rows[e * T_N + c] = t; scale[e * T_N + c] = tw[t * 2];     ++c; }
    else if (ti[t * 2 + 1] == e) { rows[e * T_N + c] = t; scale[e * T_N + c] = tw[t * 2 + 1]; ++c; }
  }
  counts[e] = c;
}

__global__ void zero_kernel(float* __restrict__ p, long n) {
  long i = (long)blockIdx.x * 256 + threadIdx.x;
  if (i < n) p[i] = 0.f;
}

__global__ void final_add_kernel(float* __restrict__ out, const float* __restrict__ a,
                                 const float* __restrict__ b, long n) {
  long i = (long)blockIdx.x * 256 + threadIdx.x;
  if (i < n) out[i] = a[i] + b[i];
}

// ---------------------------------------------------------------------------
// Host-side orchestration
// ---------------------------------------------------------------------------
static inline void gemm(hipStream_t st,
                        const float* A, long lda, long abat,
                        const float* B, long brs, long bbat,
                        float* C, long ldc, long cbat,
                        int M, int N, int K, int batch, float alpha,
                        const int* a_idx = nullptr, const int* c_idx = nullptr,
                        const float* c_scale = nullptr, const int* m_dev = nullptr,
                        int accum = 0) {
  GemmP p;
  p.A = A; p.lda = lda; p.abat = abat;
  p.B = B; p.brs = brs; p.bbat = bbat;
  p.C = C; p.ldc = ldc; p.cbat = cbat;
  p.M = M; p.N = N; p.K = K; p.alpha = alpha;
  p.a_idx = a_idx; p.c_idx = c_idx; p.c_scale = c_scale; p.m_dev = m_dev;
  p.accum = accum;
  dim3 g((N + 127) / 128, (M + 127) / 128, batch);
  gemm_bf16_wmma<<<g, 256, 0, st>>>(p);
}

extern "C" void kernel_launch(void* const* d_in, const int* in_sizes, int n_in,
                              void* d_out, int out_size, void* d_ws, size_t ws_size,
                              hipStream_t stream) {
  (void)in_sizes; (void)n_in; (void)out_size; (void)ws_size;

  const float* hidden   = (const float*)d_in[0];
  const int*   pos      = (const int*)d_in[1];
  const float* ln_in    = (const float*)d_in[2];
  const float* ln_post  = (const float*)d_in[3];
  const float* wqa      = (const float*)d_in[4];
  const float* qnorm    = (const float*)d_in[5];
  const float* wqb      = (const float*)d_in[6];
  const float* wkva     = (const float*)d_in[7];
  const float* kvnorm   = (const float*)d_in[8];
  const float* wkvb     = (const float*)d_in[9];
  const float* wo       = (const float*)d_in[10];
  const float* wgu      = (const float*)d_in[11];
  const float* wd       = (const float*)d_in[12];
  const float* router_w = (const float*)d_in[13];
  const float* moe_wg   = (const float*)d_in[14];
  const float* moe_wu   = (const float*)d_in[15];
  const float* moe_wd   = (const float*)d_in[16];
  float* out = (float*)d_out;

  char* ws = (char*)d_ws;
  size_t off = 0;
  auto alloc = [&](size_t bytes) -> char* {
    char* p = ws + off;
    off = (off + bytes + 255) & ~(size_t)255;
    return p;
  };
  float* scores = (float*)alloc((size_t)H_N * T_N * T_N * 4);
  float* res    = (float*)alloc((size_t)T_N * D_N * 4);
  float* hn     = (float*)alloc((size_t)T_N * D_N * 4);
  float* qlat   = (float*)alloc((size_t)T_N * QL_N * 4);
  float* qlatn  = (float*)alloc((size_t)T_N * QL_N * 4);
  float* q      = (float*)alloc((size_t)T_N * 3072 * 4);
  float* kv     = (float*)alloc((size_t)T_N * (KL_N + DR_N) * 4);
  float* kvcn   = (float*)alloc((size_t)T_N * KL_N * 4);
  float* kvb    = (float*)alloc((size_t)T_N * 4096 * 4);
  float* khT    = (float*)alloc((size_t)H_N * 192 * T_N * 4);
  float* o_buf  = (float*)alloc((size_t)T_N * 2048 * 4);
  float* attn   = (float*)alloc((size_t)T_N * D_N * 4);
  float* gu     = (float*)alloc((size_t)T_N * 2 * FF_N * 4);
  float* prod   = (float*)alloc((size_t)T_N * FF_N * 4);
  float* mlpout = (float*)alloc((size_t)T_N * D_N * 4);
  float* moeout = (float*)alloc((size_t)T_N * D_N * 4);
  float* logits = (float*)alloc((size_t)T_N * E_N * 4);
  int*   ti     = (int*)alloc((size_t)T_N * 2 * 4);
  float* tw     = (float*)alloc((size_t)T_N * 2 * 4);
  int*   rows   = (int*)alloc((size_t)E_N * T_N * 4);
  float* rscale = (float*)alloc((size_t)E_N * T_N * 4);
  int*   counts = (int*)alloc((size_t)E_N * 4);
  float* moe_g  = (float*)alloc((size_t)T_N * EF_N * 4);
  float* moe_u  = (float*)alloc((size_t)T_N * EF_N * 4);

  const float inv_sqrt = 1.f / sqrtf((float)(DN_N + DR_N));

  auto mla = [&](int l, const float* xin, float* aout) {
    const float* wqa_l  = wqa  + (size_t)l * D_N * QL_N;
    const float* wqb_l  = wqb  + (size_t)l * QL_N * 3072;
    const float* wkva_l = wkva + (size_t)l * D_N * (KL_N + DR_N);
    const float* wkvb_l = wkvb + (size_t)l * KL_N * 4096;
    const float* wo_l   = wo   + (size_t)l * 2048 * D_N;
    gemm(stream, xin, D_N, 0, wqa_l, QL_N, 0, qlat, QL_N, 0, T_N, QL_N, D_N, 1, 1.f);
    rms_kernel<<<T_N, 256, 0, stream>>>(qlat, QL_N, qnorm + (size_t)l * QL_N, qlatn, QL_N, QL_N);
    gemm(stream, qlatn, QL_N, 0, wqb_l, 3072, 0, q, 3072, 0, T_N, 3072, QL_N, 1, 1.f);
    gemm(stream, xin, D_N, 0, wkva_l, KL_N + DR_N, 0, kv, KL_N + DR_N, 0,
         T_N, KL_N + DR_N, D_N, 1, 1.f);
    rms_kernel<<<T_N, 256, 0, stream>>>(kv, KL_N + DR_N, kvnorm + (size_t)l * KL_N,
                                        kvcn, KL_N, KL_N);
    gemm(stream, kvcn, KL_N, 0, wkvb_l, 4096, 0, kvb, 4096, 0, T_N, 4096, KL_N, 1, 1.f);
    rope_q_kernel<<<(T_N * H_N * 32 + 255) / 256, 256, 0, stream>>>(q, pos);
    khT_kernel<<<T_N, 256, 0, stream>>>(kvb, kv, pos, khT);
    // scores[h] = (Q_h @ K_h^T) / sqrt(192)   (batched over heads)
    gemm(stream, q, 3072, 192, khT, T_N, (long)192 * T_N,
         scores, T_N, (long)T_N * T_N, T_N, T_N, 192, H_N, inv_sqrt);
    softmax_causal_kernel<<<dim3(T_N, H_N), 256, 0, stream>>>(scores, pos);
    // o[:, h*128:(h+1)*128] = P_h @ V_h
    gemm(stream, scores, T_N, (long)T_N * T_N, kvb + DN_N, 4096, 256,
         o_buf, 2048, 128, T_N, DV_N, T_N, H_N, 1.f);
    gemm(stream, o_buf, 2048, 0, wo_l, D_N, 0, aout, D_N, 0, T_N, D_N, 2048, 1, 1.f);
  };

  auto mlp = [&](int l, const float* xin, float* mout) {
    gemm(stream, xin, D_N, 0, wgu + (size_t)l * D_N * 2 * FF_N, 2 * FF_N, 0,
         gu, 2 * FF_N, 0, T_N, 2 * FF_N, D_N, 1, 1.f);
    silu_mul_gu_kernel<<<((long)T_N * FF_N + 255) / 256, 256, 0, stream>>>(gu, prod);
    gemm(stream, prod, FF_N, 0, wd + (size_t)l * FF_N * D_N, D_N, 0,
         mout, D_N, 0, T_N, D_N, FF_N, 1, 1.f);
  };

  // ---- layer 0 ----
  rms_kernel<<<T_N, 256, 0, stream>>>(hidden, D_N, ln_in, hn, D_N, D_N);
  mla(0, hn, attn);
  add_rms_kernel<<<T_N, 256, 0, stream>>>(attn, hidden, ln_post, res, hn, D_N); // res1, h1

  // ---- MoE on h1 (gathered top-2) ----
  gemm(stream, hn, D_N, 0, router_w, E_N, 0, logits, E_N, 0, T_N, E_N, D_N, 1, 1.f);
  top2_kernel<<<(T_N + 255) / 256, 256, 0, stream>>>(logits, ti, tw);
  route_build_kernel<<<1, E_N, 0, stream>>>(ti, tw, rows, rscale, counts);
  zero_kernel<<<((long)T_N * D_N + 255) / 256, 256, 0, stream>>>(moeout, (long)T_N * D_N);
  for (int e = 0; e < E_N; ++e) {
    const int* cnt = counts + e;
    gemm(stream, hn, D_N, 0, moe_wg + (size_t)e * D_N * EF_N, EF_N, 0,
         moe_g, EF_N, 0, T_N, EF_N, D_N, 1, 1.f, rows + e * T_N, nullptr, nullptr, cnt);
    gemm(stream, hn, D_N, 0, moe_wu + (size_t)e * D_N * EF_N, EF_N, 0,
         moe_u, EF_N, 0, T_N, EF_N, D_N, 1, 1.f, rows + e * T_N, nullptr, nullptr, cnt);
    silu_mul_cnt_kernel<<<((long)T_N * EF_N + 255) / 256, 256, 0, stream>>>(moe_g, moe_u, cnt);
    gemm(stream, moe_g, EF_N, 0, moe_wd + (size_t)e * EF_N * D_N, D_N, 0,
         moeout, D_N, 0, T_N, D_N, EF_N, 1, 1.f,
         nullptr, rows + e * T_N, rscale + e * T_N, cnt, /*accum=*/1);
  }

  // ---- dense MLP 0, then layer 1 ----
  mlp(0, hn, mlpout);
  add_rms_kernel<<<T_N, 256, 0, stream>>>(mlpout, res, ln_in + D_N, res, hn, D_N); // res2, h2
  mla(1, hn, attn);
  add_rms_kernel<<<T_N, 256, 0, stream>>>(attn, res, ln_post + D_N, res, hn, D_N); // res3, h3
  mlp(1, hn, mlpout);

  final_add_kernel<<<((long)T_N * D_N + 255) / 256, 256, 0, stream>>>(
      out, moeout, mlpout, (long)T_N * D_N);
}
